// CrossBallQueryFeatures_14456859919017
// MI455X (gfx1250) — matrix operand order repaired and checked by reading, not compiled, not run on hardware
//
#include <hip/hip_runtime.h>
#include <hip/hip_bf16.h>
#include <math.h>

// ---------------- types ----------------
typedef _Float16 half_t;
typedef __attribute__((ext_vector_type(16))) _Float16 v16h;
typedef __attribute__((ext_vector_type(8)))  _Float16 v8h;
typedef __attribute__((ext_vector_type(8)))  float    v8f;

// ---------------- problem constants ----------------
#define RADIUS2   0.01f
#define KNN       32
#define FPN       7
#define IN_DIM    (KNN*FPN)   /* 224 */
#define NGEO      24000
#define NTOK      8192
#define NBATCH    2
#define H1        512
#define H2        256
#define OUT_DIM   256

#define TPB_BQ    128
#define GEO_TILE  512
#define NPART     4           /* geometry partitions for ball query */
#define GEO_PART  (NGEO/NPART)

#define SENTINEL  0xFFFFFFFFFFFFFFFFull

// ---------------- weight pack: f32 [Kd,N] -> WMMA B-fragment layout f16 ----
// Packed layout: Bp[(((nt*numKt + kt)*32 + lane)*16 + j]
//   = W[kt*32 + 16*(j>>3) + 8*(lane>>4) + (j&7)][nt*16 + (lane&15)]
// so each lane's 16-half B fragment for (kt, nt) is one contiguous 32B load.
__global__ void pack_w_kernel(const float* __restrict__ W,
                              half_t* __restrict__ Bp, int N, int Kd) {
  const int p = blockIdx.x * blockDim.x + threadIdx.x;
  if (p >= Kd * N) return;
  const int numKt = Kd >> 5;
  const int j    = p & 15;
  const int lane = (p >> 4) & 31;
  const int kt   = (p >> 9) % numKt;
  const int nt   = p / (numKt << 9);
  const int k = (kt << 5) + ((j >> 3) << 4) + ((lane >> 4) << 3) + (j & 7);
  const int n = (nt << 4) + (lane & 15);
  Bp[p] = (half_t)W[(size_t)k * N + n];
}

// ---------------- ball query: partial top-32 per geometry partition --------
// one thread per token, one block-z per partition; geometry tiled through
// LDS; per-thread sorted top-32 list of (d2bits<<32 | idx) keys kept in LDS
// (stride TPB_BQ => a wave's 64-bit accesses cover all 64 banks).
__global__ __launch_bounds__(TPB_BQ)
void ballquery_partial_kernel(const float* __restrict__ geometry,
                              const float* __restrict__ tokens,
                              unsigned long long* __restrict__ partial) {
  __shared__ float sgeo[GEO_TILE * 3];
  __shared__ unsigned long long lists[TPB_BQ * KNN];

  const int tid  = threadIdx.x;
  const int b    = blockIdx.y;
  const int part = blockIdx.z;
  const int tok  = blockIdx.x * TPB_BQ + tid;

  const float* __restrict__ geo = geometry + (size_t)b * NGEO * 3;
  const size_t qoff = ((size_t)b * NTOK + tok) * 3;
  const float qx = tokens[qoff + 0];
  const float qy = tokens[qoff + 1];
  const float qz = tokens[qoff + 2];

  const int g0 = part * GEO_PART;
  const int g1 = g0 + GEO_PART;

  int cnt = 0;
  for (int base = g0; base < g1; base += GEO_TILE) {
    const int tn = (g1 - base < GEO_TILE) ? (g1 - base) : GEO_TILE;
    __syncthreads();
    for (int i = tid; i < tn * 3; i += TPB_BQ) sgeo[i] = geo[(size_t)base * 3 + i];
    __syncthreads();
    for (int j = 0; j < tn; ++j) {
      const float dx = sgeo[j * 3 + 0] - qx;
      const float dy = sgeo[j * 3 + 1] - qy;
      const float dz = sgeo[j * 3 + 2] - qz;
      const float d2 = dx * dx + dy * dy + dz * dz;
      if (d2 <= RADIUS2) {
        const unsigned long long key =
            ((unsigned long long)__float_as_uint(d2) << 32) |
            (unsigned int)(base + j);
        if (cnt < KNN || key < lists[(KNN - 1) * TPB_BQ + tid]) {
          int i2 = (cnt < KNN) ? cnt : (KNN - 1);
          while (i2 > 0 && lists[(i2 - 1) * TPB_BQ + tid] > key) {
            lists[i2 * TPB_BQ + tid] = lists[(i2 - 1) * TPB_BQ + tid];
            --i2;
          }
          lists[i2 * TPB_BQ + tid] = key;
          if (cnt < KNN) ++cnt;
        }
      }
    }
  }

  unsigned long long* __restrict__ outp =
      partial + (((size_t)part * NBATCH + b) * NTOK + tok) * KNN;
  for (int k = 0; k < KNN; ++k)
    outp[k] = (k < cnt) ? lists[k * TPB_BQ + tid] : SENTINEL;
}

// ---------------- merge partials (exact 4-way merge) + feature build -------
__global__ __launch_bounds__(256)
void merge_feats_kernel(const unsigned long long* __restrict__ partial,
                        const float* __restrict__ geometry,
                        const float* __restrict__ tokens,
                        half_t* __restrict__ feats) {
  const int t = blockIdx.x * blockDim.x + threadIdx.x;  // 0 .. B*NTOK-1
  if (t >= NBATCH * NTOK) return;
  const int b   = t / NTOK;
  const int tok = t % NTOK;

  const float* __restrict__ geo = geometry + (size_t)b * NGEO * 3;
  const size_t qoff = (size_t)t * 3;
  const float qx = tokens[qoff + 0];
  const float qy = tokens[qoff + 1];
  const float qz = tokens[qoff + 2];

  const unsigned long long* lp[NPART];
  unsigned long long heads[NPART];
  int pos[NPART];
#pragma unroll
  for (int p = 0; p < NPART; ++p) {
    lp[p] = partial + (((size_t)p * NBATCH + b) * NTOK + tok) * KNN;
    pos[p] = 0;
    heads[p] = lp[p][0];
  }

  half_t* __restrict__ f = feats + (size_t)t * IN_DIM;
  for (int k = 0; k < KNN; ++k) {
    int best = 0;
    unsigned long long bk = heads[0];
#pragma unroll
    for (int p = 1; p < NPART; ++p)
      if (heads[p] < bk) { bk = heads[p]; best = p; }

    float dist = 0.f, dxx = 0.f, dyy = 0.f, dzz = 0.f;
    if (bk != SENTINEL) {
      const unsigned int gi = (unsigned int)(bk & 0xffffffffu);
      const float gx = geo[(size_t)gi * 3 + 0];
      const float gy = geo[(size_t)gi * 3 + 1];
      const float gz = geo[(size_t)gi * 3 + 2];
      dxx = gx - qx; dyy = gy - qy; dzz = gz - qz;
      dist = sqrtf(dxx * dxx + dyy * dyy + dzz * dzz);
#pragma unroll
      for (int p = 0; p < NPART; ++p)
        if (p == best) {
          ++pos[p];
          heads[p] = (pos[p] < KNN) ? lp[p][pos[p]] : SENTINEL;
        }
    }
    f[k * FPN + 0] = (half_t)dist;
    f[k * FPN + 1] = (half_t)dxx;
    f[k * FPN + 2] = (half_t)dyy;
    f[k * FPN + 3] = (half_t)dzz;
    f[k * FPN + 4] = (half_t)dxx;
    f[k * FPN + 5] = (half_t)dyy;
    f[k * FPN + 6] = (half_t)dzz;
  }
}

// ---------------- WMMA GEMM (f16 in, f32 acc), fused bias(+gelu) ----------------
__device__ __forceinline__ float gelu_tanh(float x) {
  const float c = 0.7978845608028654f;  // sqrt(2/pi)
  return 0.5f * x * (1.f + tanhf(c * (x + 0.044715f * x * x * x)));
}

// A: [M,Kd] row-major f16. Bp: packed B fragments (see pack_w_kernel).
// One wave computes a 16x64 output tile: A fragment reused across 4 WMMAs.
template <bool GELU, bool F16OUT>
__global__ __launch_bounds__(128)
void gemm_wmma_kernel(const half_t* __restrict__ A,
                      const half_t* __restrict__ Bp,
                      const float* __restrict__ bias,
                      void* __restrict__ outp,
                      int M, int N, int Kd) {
  const int lane  = threadIdx.x & 31;
  const int wave  = blockIdx.x * 4 + (threadIdx.x >> 5);
  const int n64   = N >> 6;           // 64-wide column tiles
  const int mt    = wave / n64;
  const int nt64  = wave % n64;
  const int numKt = Kd >> 5;

  const int g   = lane >> 4;
  const int row = mt * 16 + (lane & 15);
  const half_t* __restrict__ ap0 = A + (size_t)row * Kd;

  v8f acc[4] = {};
  for (int kt = 0; kt < numKt; ++kt) {
    const int k0 = kt << 5;
    // A fragment (ISA 16-bit 16x32 layout): two contiguous 16B runs per lane.
    v8h alo = *(const v8h*)(ap0 + k0 + 8 * g);
    v8h ahi = *(const v8h*)(ap0 + k0 + 16 + 8 * g);
    v16h a;
#pragma unroll
    for (int j = 0; j < 8; ++j) { a[j] = alo[j]; a[j + 8] = ahi[j]; }
    // prefetch next A K-tile (gfx1250 global_prefetch path)
    if (kt + 1 < numKt) __builtin_prefetch(ap0 + k0 + 32 + 8 * g, 0, 1);

#pragma unroll
    for (int s = 0; s < 4; ++s) {
      const int nt = nt64 * 4 + s;
      const v16h bfrag =
          *(const v16h*)(Bp + (((size_t)nt * numKt + kt) * 32 + lane) * 16);
      acc[s] = __builtin_amdgcn_wmma_f32_16x16x32_f16(
          /*neg_a=*/false, a, /*neg_b=*/false, bfrag,
          /*c_mod=*/(short)0, acc[s], /*reuse_a=*/false, /*reuse_b=*/false);
    }
  }

#pragma unroll
  for (int s = 0; s < 4; ++s) {
    const int ncol = nt64 * 64 + s * 16 + (lane & 15);
    const float bv = bias[ncol];
#pragma unroll
    for (int r = 0; r < 8; ++r) {
      const int m = mt * 16 + r + 8 * g;  // C/D: VGPR r -> row r + 8*(lane/16)
      float v = acc[s][r] + bv;
      if (GELU) v = gelu_tanh(v);
      if (F16OUT) ((half_t*)outp)[(size_t)m * N + ncol] = (half_t)v;
      else        ((float*)outp)[(size_t)m * N + ncol]  = v;
    }
  }
}

// ---------------- launcher ----------------
extern "C" void kernel_launch(void* const* d_in, const int* in_sizes, int n_in,
                              void* d_out, int out_size, void* d_ws, size_t ws_size,
                              hipStream_t stream) {
  const float* geometry = (const float*)d_in[0];
  const float* tokens   = (const float*)d_in[1];
  const float* W1 = (const float*)d_in[2];
  const float* b1 = (const float*)d_in[3];
  const float* W2 = (const float*)d_in[4];
  const float* b2 = (const float*)d_in[5];
  const float* W3 = (const float*)d_in[6];
  const float* b3 = (const float*)d_in[7];
  float* out = (float*)d_out;
  (void)in_sizes; (void)n_in; (void)out_size; (void)ws_size;

  const int M = NBATCH * NTOK;  // 16384 tokens total

  char* ws = (char*)d_ws;
  size_t off = 0;
  half_t* W1p   = (half_t*)(ws + off); off += (size_t)IN_DIM * H1 * sizeof(half_t);
  half_t* W2p   = (half_t*)(ws + off); off += (size_t)H1 * H2 * sizeof(half_t);
  half_t* W3p   = (half_t*)(ws + off); off += (size_t)H2 * OUT_DIM * sizeof(half_t);
  half_t* feats = (half_t*)(ws + off); off += (size_t)M * IN_DIM * sizeof(half_t);
  half_t* h1    = (half_t*)(ws + off); off += (size_t)M * H1 * sizeof(half_t);
  half_t* h2    = (half_t*)(ws + off); off += (size_t)M * H2 * sizeof(half_t);
  unsigned long long* partial = (unsigned long long*)(ws + off);
  off += (size_t)NPART * M * KNN * sizeof(unsigned long long);

  // pack weights into WMMA B-fragment layout (f16)
  {
    const int n1 = IN_DIM * H1, n2 = H1 * H2, n3 = H2 * OUT_DIM;
    pack_w_kernel<<<(n1 + 255) / 256, 256, 0, stream>>>(W1, W1p, H1, IN_DIM);
    pack_w_kernel<<<(n2 + 255) / 256, 256, 0, stream>>>(W2, W2p, H2, H1);
    pack_w_kernel<<<(n3 + 255) / 256, 256, 0, stream>>>(W3, W3p, OUT_DIM, H2);
  }

  // ball query: geometry-partitioned partial top-32, then exact merge -> f16 feats
  {
    dim3 grid(NTOK / TPB_BQ, NBATCH, NPART);
    ballquery_partial_kernel<<<grid, TPB_BQ, 0, stream>>>(geometry, tokens, partial);
    merge_feats_kernel<<<(M + 255) / 256, 256, 0, stream>>>(partial, geometry, tokens, feats);
  }

  // MLP via WMMA: feats@W1+b1 -> gelu -> @W2+b2 -> gelu -> @W3+b3
  {
    const int w1 = (M / 16) * (H1 / 64);       // 8192 waves
    const int w2 = (M / 16) * (H2 / 64);       // 4096 waves
    const int w3 = (M / 16) * (OUT_DIM / 64);  // 4096 waves
    gemm_wmma_kernel<true,  true ><<<w1 / 4, 128, 0, stream>>>(feats, W1p, b1, h1, M, H1, IN_DIM);
    gemm_wmma_kernel<true,  true ><<<w2 / 4, 128, 0, stream>>>(h1,    W2p, b2, h2, M, H2, H1);
    gemm_wmma_kernel<false, false><<<w3 / 4, 128, 0, stream>>>(h2,    W3p, b3, out, M, OUT_DIM, H2);
  }
}
